// MHD_39642548142292
// MI455X (gfx1250) — matrix operand order repaired and checked by reading, not compiled, not run on hardware
//
#include <hip/hip_runtime.h>
#include <hip/hip_bf16.h>

// ---------------------------------------------------------------------------
// Problem constants (from reference)
// ---------------------------------------------------------------------------
#define FEAT 2048
#define HID  512
#define G4   2048          // 4*HID
#define BB   16
#define TT   2048
#define NOUT 22

typedef __attribute__((ext_vector_type(16))) __bf16 v16bf;
typedef __attribute__((ext_vector_type(8)))  __bf16 v8bf;
typedef __attribute__((ext_vector_type(4)))  __bf16 v4bf;
typedef __attribute__((ext_vector_type(8)))  float  v8f;

union V16U { v16bf v; v8bf h[2]; };

// Build a 16x32 bf16 A-fragment from a row-major bf16 buffer with row stride
// `ldr` (elements).  ISA layout: lane m=l&15; lanes<16 hold K {0..7,16..23},
// lanes>=16 hold K {8..15,24..31}  ->  two 16B chunks at k0 and k0+16.
static __device__ __forceinline__ v16bf load_afrag(const __bf16* base, int ldr,
                                                   int lane, int k) {
    int m  = lane & 15;
    int k8 = k + ((lane >> 4) << 3);
    const v8bf* p = (const v8bf*)(base + m * ldr + k8);
    V16U u; u.h[0] = p[0]; u.h[1] = p[2];   // +0 and +16 elements
    return u.v;
}

// Build a 32x16 bf16 B-fragment.  B[k][n] = W[n][k] (W row-major, stride ldr).
// lane n=l&15; lanes<16 hold K {0..7,16..23}, lanes>=16 hold K {8..15,24..31}
// -> two 16B chunks from row n at k + (l>=16)*8 and +16.
static __device__ __forceinline__ v16bf load_bfrag(const __bf16* Wrow, int ldr,
                                                   int lane, int k) {
    int k16 = k + ((lane >> 4) << 3);
    const v8bf* p = (const v8bf*)(Wrow + (lane & 15) * ldr + k16);
    V16U u; u.h[0] = p[0]; u.h[1] = p[2];
    return u.v;
}

static __device__ __forceinline__ float sigm(float x) {
    return 1.0f / (1.0f + __expf(-x));
}

// ---------------------------------------------------------------------------
// f32 -> bf16 conversion (grid-stride)
// ---------------------------------------------------------------------------
__global__ void k_cvt_bf16(const float* __restrict__ src, __bf16* __restrict__ dst,
                           int n) {
    for (int i = blockIdx.x * blockDim.x + threadIdx.x; i < n;
         i += gridDim.x * blockDim.x)
        dst[i] = (__bf16)src[i];
}

// fc_w [22,512] f32 -> zero-padded [32,512] bf16
__global__ void k_pad_fc(const float* __restrict__ src, __bf16* __restrict__ dst) {
    int idx = blockIdx.x * blockDim.x + threadIdx.x;   // 32*512 threads
    int row = idx >> 9, col = idx & 511;
    dst[idx] = (row < NOUT) ? (__bf16)src[row * HID + col] : (__bf16)0.0f;
}

// ---------------------------------------------------------------------------
// GEMM: xg[c] = x · w_ih[c]^T + (b_ih+b_hh)   (bf16 WMMA, fp32 accum)
// CTA tile 256(M) x 128(N), K-step 32.  8 waves, each 64x64 (4x4 WMMA tiles).
// grid = (N/128, M/256, 2 cells), block = 256.
// ---------------------------------------------------------------------------
#define LDT 40   // LDS row stride (elements) for 32-wide K tiles

__global__ __launch_bounds__(256)
void k_gemm_xg(const float* __restrict__ x,
               const __bf16* __restrict__ w0, const __bf16* __restrict__ w1,
               const float* __restrict__ bi0, const float* __restrict__ bh0,
               const float* __restrict__ bi1, const float* __restrict__ bh1,
               float* __restrict__ xg0, float* __restrict__ xg1) {
    __shared__ __bf16 As[256 * LDT];
    __shared__ __bf16 Bs[128 * LDT];

    const int cell = blockIdx.z;
    const __bf16* W  = cell ? w1  : w0;
    const float*  BI = cell ? bi1 : bi0;
    const float*  BH = cell ? bh1 : bh0;
    float*        XG = cell ? xg1 : xg0;

    const int tid  = threadIdx.x;
    const int lane = tid & 31;
    const int w    = tid >> 5;
    const int wm   = w & 3;        // 4 wave-rows   (64 rows each)
    const int wn   = w >> 2;       // 2 wave-cols   (64 cols each)
    const int m0   = blockIdx.y * 256;
    const int n0   = blockIdx.x * 128;

    v8f acc[4][4];
#pragma unroll
    for (int i = 0; i < 4; ++i)
#pragma unroll
        for (int j = 0; j < 4; ++j) acc[i][j] = {};

    for (int kk = 0; kk < FEAT; kk += 32) {
        // ---- stage A tile (256x32), convert f32 -> bf16 ----
        {
            const float* src = x + (size_t)(m0 + tid) * FEAT + kk;
            __bf16* dst = As + tid * LDT;
#pragma unroll
            for (int c = 0; c < 8; ++c) {
                float4 v = *(const float4*)(src + c * 4);
                v4bf p; p[0] = (__bf16)v.x; p[1] = (__bf16)v.y;
                        p[2] = (__bf16)v.z; p[3] = (__bf16)v.w;
                *(v4bf*)(dst + c * 4) = p;
            }
        }
        // ---- stage B tile (128x32) from bf16 weights ----
        {
            int r  = tid >> 1;
            int c0 = (tid & 1) * 16;
            const uint4* s = (const uint4*)(W + (size_t)(n0 + r) * FEAT + kk + c0);
            uint4* d = (uint4*)(Bs + r * LDT + c0);
            d[0] = s[0]; d[1] = s[1];
        }
        __syncthreads();

        v16bf bf[4];
#pragma unroll
        for (int nt = 0; nt < 4; ++nt)
            bf[nt] = load_bfrag(Bs + (wn * 64 + nt * 16) * LDT, LDT, lane, 0);
#pragma unroll
        for (int mt = 0; mt < 4; ++mt) {
            v16bf af = load_afrag(As + (wm * 64 + mt * 16) * LDT, LDT, lane, 0);
#pragma unroll
            for (int nt = 0; nt < 4; ++nt)
                acc[mt][nt] = __builtin_amdgcn_wmma_f32_16x16x32_bf16(
                    false, af, false, bf[nt], (short)0, acc[mt][nt], false, false);
        }
        __syncthreads();
    }

    // ---- epilogue: add (b_ih+b_hh), store D tiles ----
#pragma unroll
    for (int nt = 0; nt < 4; ++nt) {
        int col = n0 + wn * 64 + nt * 16 + (lane & 15);
        float bias = BI[col] + BH[col];
#pragma unroll
        for (int mt = 0; mt < 4; ++mt) {
            int rbase = m0 + wm * 64 + mt * 16 + ((lane >> 4) << 3);
#pragma unroll
            for (int r = 0; r < 8; ++r)
                XG[(size_t)(rbase + r) * G4 + col] = acc[mt][nt][r] + bias;
        }
    }
}

// ---------------------------------------------------------------------------
// Recurrent LSTM + fused head.  grid = 2 (cell), block = 544 (17 waves).
// Waves 0..15: hidden slice of 32 cols each -> 8 WMMA tiles (4 gates x 2),
//              c-state lives in registers, h written to LDS as bf16.
// Wave 16:     score head for h_{t-1} via 2 padded WMMA tiles (OUT=22 -> 32).
// ---------------------------------------------------------------------------
#define HSTR 520   // LDS row stride for h (bf16 elements)

__global__ __launch_bounds__(544)
void k_lstm(const float* __restrict__ xg0, const float* __restrict__ xg1,
            const __bf16* __restrict__ whh0, const __bf16* __restrict__ whh1,
            const __bf16* __restrict__ fcw0, const __bf16* __restrict__ fcw1,
            const float* __restrict__ fcb0, const float* __restrict__ fcb1,
            float* __restrict__ outbase) {
    __shared__ __bf16 hl[BB * HSTR];

    const int cell = blockIdx.x;
    const float*  XG  = cell ? xg1  : xg0;
    const __bf16* WHH = cell ? whh1 : whh0;
    const __bf16* FCW = cell ? fcw1 : fcw0;
    const float*  FCB = cell ? fcb1 : fcb0;
    float* OUT = outbase + (size_t)cell * BB * TT * NOUT;

    const int tid  = threadIdx.x;
    const int lane = tid & 31;
    const int w    = tid >> 5;

    // zero h_0
    for (int i = tid; i < BB * HSTR; i += blockDim.x) hl[i] = (__bf16)0.0f;
    __syncthreads();

    v8f acc[8];          // gate pre-activations (4 gates x 2 half-tiles)
    v8f cc[2];  cc[0] = {}; cc[1] = {};     // cell state (registers)
    v8f hr[2];  hr[0] = {}; hr[1] = {};
    v8f hacc[2];

    const int rrow = (lane >> 4) << 3;      // 0 or 8 (D-tile row base)
    const int jcol = (w < 16) ? (w * 32) : 0;

    for (int t = 0; t <= TT; ++t) {
        const bool doGate = (w < 16) && (t < TT);
        const bool doHead = (w == 16) && (t > 0);

        if (doGate) {
            // init accumulators from xg[:, t, :]   (bias already folded in)
#pragma unroll
            for (int a = 0; a < 8; ++a) {
                int col = (a >> 1) * HID + jcol + (a & 1) * 16 + (lane & 15);
                const float* p = XG + (size_t)rrow * TT * G4 + (size_t)t * G4 + col;
#pragma unroll
                for (int r = 0; r < 8; ++r)
                    acc[a][r] = p[(size_t)r * TT * G4];
            }
            // prefetch next timestep's xg slice for this wave
            if (t + 1 < TT) {
#pragma unroll
                for (int a = 0; a < 8; a += 2) {
                    const float* p = XG + ((size_t)(lane & 15) * TT + (t + 1)) * G4
                                        + (a >> 1) * HID + jcol;
                    __builtin_prefetch(p, 0, 3);
                }
            }
            // h_{t-1} · w_hh^T
            for (int ks = 0; ks < HID / 32; ++ks) {
                v16bf af = load_afrag(hl, HSTR, lane, ks * 32);
#pragma unroll
                for (int a = 0; a < 8; ++a) {
                    int nb = (a >> 1) * HID + jcol + (a & 1) * 16;
                    v16bf bf = load_bfrag(WHH + (size_t)nb * HID, HID, lane, ks * 32);
                    acc[a] = __builtin_amdgcn_wmma_f32_16x16x32_bf16(
                        false, af, false, bf, (short)0, acc[a], false, false);
                }
            }
            // element-wise gates: i=acc[0..1] f=acc[2..3] g=acc[4..5] o=acc[6..7]
#pragma unroll
            for (int h = 0; h < 2; ++h)
#pragma unroll
                for (int r = 0; r < 8; ++r) {
                    float gi = acc[0 + h][r], gf = acc[2 + h][r];
                    float gg = acc[4 + h][r], go = acc[6 + h][r];
                    float cn = sigm(gf) * cc[h][r] + sigm(gi) * tanhf(gg);
                    cc[h][r] = cn;
                    hr[h][r] = sigm(go) * tanhf(cn);
                }
        }

        if (doHead) {
            hacc[0] = {}; hacc[1] = {};
            for (int ks = 0; ks < HID / 32; ++ks) {
                v16bf af = load_afrag(hl, HSTR, lane, ks * 32);
#pragma unroll
                for (int nt = 0; nt < 2; ++nt) {
                    v16bf bf = load_bfrag(FCW + (size_t)(nt * 16) * HID, HID,
                                          lane, ks * 32);
                    hacc[nt] = __builtin_amdgcn_wmma_f32_16x16x32_bf16(
                        false, af, false, bf, (short)0, hacc[nt], false, false);
                }
            }
            int tt = t - 1;
#pragma unroll
            for (int nt = 0; nt < 2; ++nt) {
                int col = nt * 16 + (lane & 15);
                if (col < NOUT) {
                    float bias = FCB[col];
#pragma unroll
                    for (int r = 0; r < 8; ++r)
                        OUT[((size_t)(rrow + r) * TT + tt) * NOUT + col] =
                            hacc[nt][r] + bias;
                }
            }
        }

        __syncthreads();   // all h_lds reads of this step complete

        if (doGate) {
#pragma unroll
            for (int h = 0; h < 2; ++h)
#pragma unroll
                for (int r = 0; r < 8; ++r)
                    hl[(rrow + r) * HSTR + jcol + h * 16 + (lane & 15)] =
                        (__bf16)hr[h][r];
        }

        __syncthreads();   // h_t visible for next iteration
    }
}

// ---------------------------------------------------------------------------
// Launch
// ---------------------------------------------------------------------------
extern "C" void kernel_launch(void* const* d_in, const int* in_sizes, int n_in,
                              void* d_out, int out_size, void* d_ws, size_t ws_size,
                              hipStream_t stream) {
    const float* x       = (const float*)d_in[0];
    const float* w_ih    = (const float*)d_in[1];
    const float* w_hh    = (const float*)d_in[2];
    const float* b_ih    = (const float*)d_in[3];
    const float* b_hh    = (const float*)d_in[4];
    const float* w_ih_ed = (const float*)d_in[5];
    const float* w_hh_ed = (const float*)d_in[6];
    const float* b_ih_ed = (const float*)d_in[7];
    const float* b_hh_ed = (const float*)d_in[8];
    const float* fc_w    = (const float*)d_in[9];
    const float* fc_b    = (const float*)d_in[10];
    const float* fc_ed_w = (const float*)d_in[11];
    const float* fc_ed_b = (const float*)d_in[12];

    char* ws = (char*)d_ws;
    size_t off = 0;
    float* xg0 = (float*)(ws + off);  off += (size_t)BB * TT * G4 * 4;
    float* xg1 = (float*)(ws + off);  off += (size_t)BB * TT * G4 * 4;
    __bf16* wih0 = (__bf16*)(ws + off); off += (size_t)G4 * FEAT * 2;
    __bf16* wih1 = (__bf16*)(ws + off); off += (size_t)G4 * FEAT * 2;
    __bf16* whh0 = (__bf16*)(ws + off); off += (size_t)G4 * HID * 2;
    __bf16* whh1 = (__bf16*)(ws + off); off += (size_t)G4 * HID * 2;
    __bf16* fcw0 = (__bf16*)(ws + off); off += (size_t)32 * HID * 2;
    __bf16* fcw1 = (__bf16*)(ws + off); off += (size_t)32 * HID * 2;

    // weight conversions to bf16
    k_cvt_bf16<<<2048, 256, 0, stream>>>(w_ih,    wih0, G4 * FEAT);
    k_cvt_bf16<<<2048, 256, 0, stream>>>(w_ih_ed, wih1, G4 * FEAT);
    k_cvt_bf16<<<1024, 256, 0, stream>>>(w_hh,    whh0, G4 * HID);
    k_cvt_bf16<<<1024, 256, 0, stream>>>(w_hh_ed, whh1, G4 * HID);
    k_pad_fc<<<64, 256, 0, stream>>>(fc_w,    fcw0);
    k_pad_fc<<<64, 256, 0, stream>>>(fc_ed_w, fcw1);

    // big input-projection GEMM for both cells
    k_gemm_xg<<<dim3(G4 / 128, (BB * TT) / 256, 2), 256, 0, stream>>>(
        x, wih0, wih1, b_ih, b_hh, b_ih_ed, b_hh_ed, xg0, xg1);

    // sequential recurrence + fused heads
    k_lstm<<<dim3(2), dim3(544), 0, stream>>>(
        xg0, xg1, whh0, whh1, fcw0, fcw1, fc_b, fc_ed_b, (float*)d_out);
}
